// RGCN_58420145160624
// MI455X (gfx1250) — compile-verified, matrix operand order checked by reading, not run on hardware
//
#include <hip/hip_runtime.h>
#include <hip/hip_bf16.h>

// ---------------------------------------------------------------------------
// Types for CDNA5 WMMA (wave32): v_wmma_f32_16x16x32_bf16
// ---------------------------------------------------------------------------
typedef __bf16 bf16_t;
typedef bf16_t bf16x16 __attribute__((ext_vector_type(16)));
typedef float  f32x8   __attribute__((ext_vector_type(8)));

__device__ __forceinline__ float sigf(float x) { return 1.0f / (1.0f + expf(-x)); }

// ---------------------------------------------------------------------------
// Generic bf16 WMMA GEMM:  C = act( A(MxK) * B(KxN) + Cin + bias )
//   AF32: A is fp32 row-major (converted on the fly) vs bf16 row-major
//   CIN : accumulate fp32 Cin (stride ldci)
//   BIAS: add bias[n]
//   RELU: apply max(0,x)
//   CF  : write fp32 output (stride ldc);  CBF: write bf16 output (ldcb)
//   B: pre-transposed bf16, Bt is N x K row-major (lane loads 16 contiguous K)
// Wave tile 32(M) x 64(N): 2 A frags x 4 B frags -> 8 WMMA per K-step.
// Requires M%32==0, N%64==0, K%32==0. EXEC stays all-ones (wave-uniform flow).
// ---------------------------------------------------------------------------
template <bool AF32, bool CIN, bool BIAS, bool RELU, bool CF, bool CBF>
__global__ __launch_bounds__(256) void wmma_gemm_bf16_kernel(
    const __hip_bfloat16* __restrict__ Abf, const float* __restrict__ A32,
    const __hip_bfloat16* __restrict__ Bt,
    const float* __restrict__ bias,
    const float* __restrict__ Cin, int ldci,
    float* __restrict__ Cf, int ldc,
    __hip_bfloat16* __restrict__ Cbf, int ldcb,
    int M, int N, int K)
{
  const int lane  = threadIdx.x & 31;
  const int wid   = blockIdx.x * (blockDim.x >> 5) + (threadIdx.x >> 5);
  const int nwave = gridDim.x * (blockDim.x >> 5);
  const int tilesN = N >> 6;                 // 64-wide wave tiles in N
  const int tiles  = (M >> 5) * tilesN;
  const int rsel = lane & 15;                // row select within 16-row block
  const int ksel = (lane >> 4) << 4;         // K sub-block: 0 or 16
  const int nco  = lane & 15;                // C: column within 16
  const int mro  = (lane >> 4) << 3;         // C: row offset 0 or 8

  for (int t = wid; t < tiles; t += nwave) {
    const int mt = (t / tilesN) << 5;
    const int nt = (t % tilesN) << 6;

    f32x8 c[2][4];
#pragma unroll
    for (int im = 0; im < 2; ++im)
#pragma unroll
      for (int jn = 0; jn < 4; ++jn) c[im][jn] = (f32x8){};

    const float*          pa0f = AF32 ? (A32 + (size_t)(mt + rsel)      * K + ksel) : nullptr;
    const float*          pa1f = AF32 ? (A32 + (size_t)(mt + 16 + rsel) * K + ksel) : nullptr;
    const __hip_bfloat16* pa0b = AF32 ? nullptr : (Abf + (size_t)(mt + rsel)      * K + ksel);
    const __hip_bfloat16* pa1b = AF32 ? nullptr : (Abf + (size_t)(mt + 16 + rsel) * K + ksel);
    const __hip_bfloat16* pb[4];
#pragma unroll
    for (int jn = 0; jn < 4; ++jn) pb[jn] = Bt + (size_t)(nt + jn * 16 + rsel) * K + ksel;

    for (int k0 = 0; k0 < K; k0 += 32) {
      bf16x16 a0, a1;
      if (AF32) {
#pragma unroll
        for (int i = 0; i < 16; ++i) { a0[i] = (bf16_t)pa0f[k0 + i]; a1[i] = (bf16_t)pa1f[k0 + i]; }
      } else {
        a0 = *reinterpret_cast<const bf16x16*>(pa0b + k0);
        a1 = *reinterpret_cast<const bf16x16*>(pa1b + k0);
      }
      bf16x16 b[4];
#pragma unroll
      for (int jn = 0; jn < 4; ++jn) {
        b[jn] = *reinterpret_cast<const bf16x16*>(pb[jn] + k0);
        __builtin_prefetch(pb[jn] + k0 + 32, 0, 1);  // speculative: dropped if OOB
      }
#pragma unroll
      for (int jn = 0; jn < 4; ++jn) {
        c[0][jn] = __builtin_amdgcn_wmma_f32_16x16x32_bf16(false, a0, false, b[jn], (short)0, c[0][jn], false, false);
        c[1][jn] = __builtin_amdgcn_wmma_f32_16x16x32_bf16(false, a1, false, b[jn], (short)0, c[1][jn], false, false);
      }
    }

    // Branch-free epilogue (flags are compile-time): bulk Cin loads per 16x16
    // sub-tile so the backend can clause them under a single wait.
#pragma unroll
    for (int im = 0; im < 2; ++im)
#pragma unroll
      for (int jn = 0; jn < 4; ++jn) {
        const int bm = mt + (im << 4);
        const int bn = nt + (jn << 4);
        const int m0 = bm + mro;
        const int n  = bn + nco;
        float bv = 0.0f;
        if (BIAS) bv = bias[n];
        float cinv[8];
        if (CIN) {
#pragma unroll
          for (int r = 0; r < 8; ++r) cinv[r] = Cin[(size_t)(m0 + r) * ldci + n];
        }
#pragma unroll
        for (int r = 0; r < 8; ++r) {
          float v = c[im][jn][r];
          if (CIN)  v += cinv[r];
          if (BIAS) v += bv;
          if (RELU) v = fmaxf(v, 0.0f);
          if (CF)  Cf[(size_t)(m0 + r) * ldc + n] = v;
          if (CBF) Cbf[(size_t)(m0 + r) * ldcb + n] = __float2bfloat16(v);
        }
      }
  }
}

// ---------------------------------------------------------------------------
// Small helper kernels
// ---------------------------------------------------------------------------
// Transpose + convert: Bt[n*K + k] = bf16(W[k*N + n])
__global__ void transpose_cvt_kernel(const float* __restrict__ W,
                                     __hip_bfloat16* __restrict__ Bt, int K, int N) {
  int idx = blockIdx.x * blockDim.x + threadIdx.x;
  if (idx >= K * N) return;
  int k = idx / N, n = idx % N;
  Bt[(size_t)n * K + k] = __float2bfloat16(W[idx]);
}

// dst_bf16[i] = bf16(src[i] * scale)
__global__ void cvt_scale_kernel(__hip_bfloat16* __restrict__ dst,
                                 const float* __restrict__ src, int n, float scale) {
  int i = blockIdx.x * blockDim.x + threadIdx.x;
  if (i < n) dst[i] = __float2bfloat16(src[i] * scale);
}

// broadcast one bf16 row into every row of dst (stride ldd)
__global__ void bcast_rows_kernel(__hip_bfloat16* __restrict__ dst, int ldd,
                                  const __hip_bfloat16* __restrict__ src, int cols, int rows) {
  long long i = (long long)blockIdx.x * blockDim.x + threadIdx.x;
  long long total = (long long)rows * cols;
  if (i >= total) return;
  int r = (int)(i / cols), c = (int)(i % cols);
  dst[(size_t)r * ldd + c] = src[c];
}

// 2-D strided bf16 copy
__global__ void copy2d_kernel(__hip_bfloat16* __restrict__ dst, int ldd,
                              const __hip_bfloat16* __restrict__ src, int lds,
                              int cols, int rows) {
  long long i = (long long)blockIdx.x * blockDim.x + threadIdx.x;
  long long total = (long long)rows * cols;
  if (i >= total) return;
  int r = (int)(i / cols), c = (int)(i % cols);
  dst[(size_t)r * ldd + c] = src[(size_t)r * lds + c];
}

// LSTM gate GEMV: one wave per gate row. g[r] = dot(x, W[r,:]) + bih[r] + bhh[r]
__global__ void lstm_gates_kernel(const float* __restrict__ x, const float* __restrict__ W,
                                  const float* __restrict__ bih, const float* __restrict__ bhh,
                                  float* __restrict__ g, int G, int K) {
  int lane = threadIdx.x & 31;
  int wid  = blockIdx.x * (blockDim.x >> 5) + (threadIdx.x >> 5);
  if (wid >= G) return;
  const float* w = W + (size_t)wid * K;
  float s = 0.0f;
  for (int k = lane; k < K; k += 32) s += x[k] * w[k];
#pragma unroll
  for (int off = 16; off > 0; off >>= 1) s += __shfl_xor(s, off, 32);
  if (lane == 0) g[wid] = s + bih[wid] + bhh[wid];
}

// LSTM cell (zero h0/c0), torch gate order i,f,c,o: h = sig(o)*tanh(sig(i)*tanh(c))
// Output layout: [forward(Hu), backward(Hu)]; gate buffer g: [dir0 4*Hu][dir1 4*Hu]
__global__ void lstm_combine_kernel(const float* __restrict__ g, float* __restrict__ h, int Hu) {
  int j = blockIdx.x * blockDim.x + threadIdx.x;
  if (j >= 2 * Hu) return;
  int d = j / Hu, u = j % Hu;
  const float* gb = g + (size_t)d * 4 * Hu;
  float gi = gb[u], gc = gb[2 * Hu + u], go = gb[3 * Hu + u];
  float cc = sigf(gi) * tanhf(gc);
  h[j] = sigf(go) * tanhf(cc);
}

// Edge scatter-add: agg[row[e], :] += H[col[e], :]  (512-wide, float4 per thread)
__global__ void scatter_add_kernel(const float* __restrict__ H, const int* __restrict__ rows,
                                   const int* __restrict__ cols, float* __restrict__ agg, int E) {
  long long total = (long long)E * 128;
  long long stride = (long long)gridDim.x * blockDim.x;
  for (long long i = (long long)blockIdx.x * blockDim.x + threadIdx.x; i < total; i += stride) {
    int e  = (int)(i >> 7);
    int c4 = ((int)i & 127) << 2;
    int r = rows[e], c = cols[e];
    const float4 v = *reinterpret_cast<const float4*>(H + (size_t)c * 512 + c4);
    float* dst = agg + (size_t)r * 512 + c4;
    atomicAdd(dst + 0, v.x); atomicAdd(dst + 1, v.y);
    atomicAdd(dst + 2, v.z); atomicAdd(dst + 3, v.w);
  }
}

// H = tanh(U)*sig(G) + H*(1-sig(G));  also refresh bf16 copy of H
__global__ void gate_update_kernel(float* __restrict__ H, const float* __restrict__ U,
                                   const float* __restrict__ G, __hip_bfloat16* __restrict__ Hbf,
                                   long long n) {
  long long i = (long long)blockIdx.x * blockDim.x + threadIdx.x;
  if (i >= n) return;
  float s  = sigf(G[i]);
  float hn = tanhf(U[i]) * s + H[i] * (1.0f - s);
  H[i] = hn;
  Hbf[i] = __float2bfloat16(hn);
}

// scores[row] = dot(o2f[row,:128], o3w) + o3b
__global__ void scores_kernel(const float* __restrict__ o2f, const float* __restrict__ o3w,
                              const float* __restrict__ o3b, float* __restrict__ scores, int M) {
  int row = blockIdx.x * blockDim.x + threadIdx.x;
  if (row >= M) return;
  const float* p = o2f + (size_t)row * 128;
  float s = o3b[0];
#pragma unroll 8
  for (int j = 0; j < 128; ++j) s += p[j] * o3w[j];
  scores[row] = s;
}

// out[c] = max_k scores[cand[c*16+k]]
__global__ void candmax_kernel(const float* __restrict__ scores, const int* __restrict__ cand,
                               float* __restrict__ out, int C) {
  int c = blockIdx.x * blockDim.x + threadIdx.x;
  if (c >= C) return;
  float m = -3.4e38f;
  for (int k = 0; k < 16; ++k) m = fmaxf(m, scores[cand[c * 16 + k]]);
  out[c] = m;
}

// ---------------------------------------------------------------------------
// Host orchestration
// ---------------------------------------------------------------------------
static inline int gemm_blocks(int m, int n) {
  int tiles = (m / 32) * (n / 64);
  return (tiles + 7) / 8;                     // 8 waves / block
}

extern "C" void kernel_launch(void* const* d_in, const int* in_sizes, int n_in,
                              void* d_out, int out_size, void* d_ws, size_t ws_size,
                              hipStream_t stream) {
  // Inputs (setup_inputs dict order; params flattened in insertion order)
  const float* nodes    = (const float*)d_in[0];   // (M,3072)
  const float* q        = (const float*)d_in[1];   // (1,3072)
  const float* l1Wih_f  = (const float*)d_in[2];
  const float* l1bih_f  = (const float*)d_in[3];
  const float* l1bhh_f  = (const float*)d_in[4];
  const float* l1Wih_b  = (const float*)d_in[5];
  const float* l1bih_b  = (const float*)d_in[6];
  const float* l1bhh_b  = (const float*)d_in[7];
  const float* l2Wih_f  = (const float*)d_in[8];
  const float* l2bih_f  = (const float*)d_in[9];
  const float* l2bhh_f  = (const float*)d_in[10];
  const float* l2Wih_b  = (const float*)d_in[11];
  const float* l2bih_b  = (const float*)d_in[12];
  const float* l2bhh_b  = (const float*)d_in[13];
  const float* nfc_W    = (const float*)d_in[14];  // (3072,256)
  const float* nfc_b    = (const float*)d_in[15];
  const float* qn1_W    = (const float*)d_in[16];  // (512,1024)
  const float* qn1_b    = (const float*)d_in[17];
  const float* qn2_W    = (const float*)d_in[18];  // (1024,512)
  const float* qn2_b    = (const float*)d_in[19];
  const float* W0       = (const float*)d_in[20];  // (512,512)
  const float* Wr       = (const float*)d_in[21];  // (4,512,512)
  const float* Wg       = (const float*)d_in[22];  // (1024,512)
  const float* o1_W     = (const float*)d_in[23];  // (768,256)
  const float* o1_b     = (const float*)d_in[24];
  const float* o2_W     = (const float*)d_in[25];  // (256,128)
  const float* o2_b     = (const float*)d_in[26];
  const float* o3_W     = (const float*)d_in[27];  // (128,)
  const float* o3_b     = (const float*)d_in[28];  // (1,)
  const int*   edge_idx = (const int*)d_in[29];    // (4,2,E)
  const int*   cand_idx = (const int*)d_in[30];    // (C,16)

  const int M = in_sizes[0] / 3072;                // 20000 (multiple of 32)
  const int E = in_sizes[29] / (4 * 2);            // 320000
  const int C = out_size;                          // 64
  float* out = (float*)d_out;
  const float scale = 1.0f / (float)(M - 1);

  // ---- workspace layout (256B aligned) ----
  char* ws = (char*)d_ws;
  size_t off = 0;
  auto alloc = [&](size_t bytes) -> char* {
    char* p = ws + off;
    off += (bytes + 255) & ~(size_t)255;
    return p;
  };
  float* H   = (float*)alloc((size_t)M * 512 * 4);   // node state (fp32)
  float* U   = (float*)alloc((size_t)M * 512 * 4);   // msgs / update accumulator
  float* G   = (float*)alloc((size_t)M * 512 * 4);   // agg scratch / gate preact
  float* QS  = (float*)alloc(4096 * 4);              // query-encoder scratch
  __hip_bfloat16* X     = (__hip_bfloat16*)alloc((size_t)M * 512 * 2);   // [qrep|ne]
  __hip_bfloat16* B1024 = (__hip_bfloat16*)alloc((size_t)M * 1024 * 2);  // T1 / gate-in / XO
  __hip_bfloat16* HBF   = (__hip_bfloat16*)alloc((size_t)M * 512 * 2);
  __hip_bfloat16* AGGBF = (__hip_bfloat16*)alloc((size_t)M * 512 * 2);   // also o1 out
  __hip_bfloat16* QREPBF= (__hip_bfloat16*)alloc(256 * 2);
  __hip_bfloat16* nfcT  = (__hip_bfloat16*)alloc((size_t)3072 * 256 * 2);
  __hip_bfloat16* qn1T  = (__hip_bfloat16*)alloc((size_t)512 * 1024 * 2);
  __hip_bfloat16* qn2T  = (__hip_bfloat16*)alloc((size_t)1024 * 512 * 2);
  __hip_bfloat16* W0T   = (__hip_bfloat16*)alloc((size_t)512 * 512 * 2);
  __hip_bfloat16* WrT   = (__hip_bfloat16*)alloc((size_t)4 * 512 * 512 * 2);
  __hip_bfloat16* WgT   = (__hip_bfloat16*)alloc((size_t)1024 * 512 * 2);
  __hip_bfloat16* o1T   = (__hip_bfloat16*)alloc((size_t)768 * 256 * 2);
  __hip_bfloat16* o2T   = (__hip_bfloat16*)alloc((size_t)256 * 128 * 2);
  (void)ws_size; (void)n_in;
  // aliases (phases do not overlap)
  float* qg1   = QS;            // 2048 l1 gates
  float* h1    = QS + 2048;     // 512
  float* qg2   = QS + 2560;     // 1024 l2 gates
  float* qrepf = QS + 3584;     // 256
  float* O2F    = U;            // (M,128) fp32, after layers
  float* SCORES = G;            // (M,) fp32, after layers
  __hip_bfloat16* XO   = B1024; // (M,768) bf16, after layers
  __hip_bfloat16* O1BF = AGGBF; // (M,256) bf16, after layers

  auto tcvt = [&](const float* W, __hip_bfloat16* T, int k, int n) {
    int tot = k * n;
    transpose_cvt_kernel<<<(tot + 255) / 256, 256, 0, stream>>>(W, T, k, n);
  };

  // ---- 1) weight transpose + bf16 convert ----
  tcvt(nfc_W, nfcT, 3072, 256);
  tcvt(qn1_W, qn1T, 512, 1024);
  tcvt(qn2_W, qn2T, 1024, 512);
  tcvt(W0, W0T, 512, 512);
  for (int r = 0; r < 4; ++r) tcvt(Wr + (size_t)r * 512 * 512, WrT + (size_t)r * 512 * 512, 512, 512);
  tcvt(Wg, WgT, 1024, 512);
  tcvt(o1_W, o1T, 768, 256);
  tcvt(o2_W, o2T, 256, 128);

  // ---- 2) query encoder (BiLSTM x2, seq_len==1) ----
  lstm_gates_kernel<<<(1024 * 32) / 256, 256, 0, stream>>>(q, l1Wih_f, l1bih_f, l1bhh_f, qg1, 1024, 3072);
  lstm_gates_kernel<<<(1024 * 32) / 256, 256, 0, stream>>>(q, l1Wih_b, l1bih_b, l1bhh_b, qg1 + 1024, 1024, 3072);
  lstm_combine_kernel<<<2, 256, 0, stream>>>(qg1, h1, 256);
  lstm_gates_kernel<<<(512 * 32) / 256, 256, 0, stream>>>(h1, l2Wih_f, l2bih_f, l2bhh_f, qg2, 512, 512);
  lstm_gates_kernel<<<(512 * 32) / 256, 256, 0, stream>>>(h1, l2Wih_b, l2bih_b, l2bhh_b, qg2 + 512, 512, 512);
  lstm_combine_kernel<<<1, 256, 0, stream>>>(qg2, qrepf, 128);
  cvt_scale_kernel<<<1, 256, 0, stream>>>(QREPBF, qrepf, 256, 1.0f);

  // ---- 3) X = [qrep | relu(nodes@nfc_W + b)]  (M,512) bf16 ----
  {
    long long tot = (long long)M * 256;
    bcast_rows_kernel<<<(int)((tot + 255) / 256), 256, 0, stream>>>(X, 512, QREPBF, 256, M);
  }
  // nfc: A=fp32 nodes, bias+relu, bf16 out only
  wmma_gemm_bf16_kernel<true, false, true, true, false, true>
      <<<gemm_blocks(M, 256), 256, 0, stream>>>(nullptr, nodes, nfcT, nfc_b,
                                                nullptr, 0, nullptr, 0, X + 256, 512, M, 256, 3072);

  // ---- 4) H = relu(relu(X@qn1 + b)@qn2 + b) ----
  wmma_gemm_bf16_kernel<false, false, true, true, false, true>
      <<<gemm_blocks(M, 1024), 256, 0, stream>>>(X, nullptr, qn1T, qn1_b,
                                                 nullptr, 0, nullptr, 0, B1024, 1024, M, 1024, 512);
  wmma_gemm_bf16_kernel<false, false, true, true, true, true>
      <<<gemm_blocks(M, 512), 256, 0, stream>>>(B1024, nullptr, qn2T, qn2_b,
                                                nullptr, 0, H, 512, HBF, 512, M, 512, 1024);

  // ---- 5) 3 weight-shared gated RGCN hops ----
  const size_t hb = (size_t)M * 512 * 4;
  for (int layer = 0; layer < 3; ++layer) {
    hipMemsetAsync(U, 0, hb, stream);                       // msgs = 0
    for (int r = 0; r < 4; ++r) {
      hipMemsetAsync(G, 0, hb, stream);                     // agg = 0
      const int* rows = edge_idx + (size_t)r * 2 * E;
      const int* cols = rows + E;
      long long tot = (long long)E * 128;
      int blocks = (int)((tot + 255) / 256);
      scatter_add_kernel<<<blocks, 256, 0, stream>>>(H, rows, cols, G, E);
      int ne = M * 512;
      cvt_scale_kernel<<<(ne + 255) / 256, 256, 0, stream>>>(AGGBF, G, ne, scale);
      // msgs += agg @ Wr[r]   (accumulate fp32, no bias/act)
      wmma_gemm_bf16_kernel<false, true, false, false, true, false>
          <<<gemm_blocks(M, 512), 256, 0, stream>>>(AGGBF, nullptr, WrT + (size_t)r * 512 * 512,
                                                    nullptr, U, 512, U, 512, nullptr, 0, M, 512, 512);
    }
    // update = H@W0 + msgs  (fp32 in U, bf16 copy into gate-input cols [0,512))
    wmma_gemm_bf16_kernel<false, true, false, false, true, true>
        <<<gemm_blocks(M, 512), 256, 0, stream>>>(HBF, nullptr, W0T, nullptr,
                                                  U, 512, U, 512, B1024, 1024, M, 512, 512);
    {
      long long tot = (long long)M * 512;
      copy2d_kernel<<<(int)((tot + 255) / 256), 256, 0, stream>>>(B1024 + 512, 1024, HBF, 512, 512, M);
    }
    // gate preact = [update|H] @ Wg
    wmma_gemm_bf16_kernel<false, false, false, false, true, false>
        <<<gemm_blocks(M, 512), 256, 0, stream>>>(B1024, nullptr, WgT, nullptr,
                                                  nullptr, 0, G, 512, nullptr, 0, M, 512, 1024);
    {
      long long tot = (long long)M * 512;
      gate_update_kernel<<<(int)((tot + 255) / 256), 256, 0, stream>>>(H, U, G, HBF, tot);
    }
  }

  // ---- 6) output MLP + candidate max-pool ----
  {
    long long tot = (long long)M * 256;
    bcast_rows_kernel<<<(int)((tot + 255) / 256), 256, 0, stream>>>(XO, 768, QREPBF, 256, M);
    tot = (long long)M * 512;
    copy2d_kernel<<<(int)((tot + 255) / 256), 256, 0, stream>>>(XO + 256, 768, HBF, 512, 512, M);
  }
  wmma_gemm_bf16_kernel<false, false, true, true, false, true>
      <<<gemm_blocks(M, 256), 256, 0, stream>>>(XO, nullptr, o1T, o1_b,
                                                nullptr, 0, nullptr, 0, O1BF, 256, M, 256, 768);
  wmma_gemm_bf16_kernel<false, false, true, true, true, false>
      <<<gemm_blocks(M, 128), 256, 0, stream>>>(O1BF, nullptr, o2T, o2_b,
                                                nullptr, 0, O2F, 128, nullptr, 0, M, 128, 256);
  scores_kernel<<<(M + 255) / 256, 256, 0, stream>>>(O2F, o3_W, o3_b, SCORES, M);
  candmax_kernel<<<1, 64, 0, stream>>>(SCORES, cand_idx, out, C);
}
